// HSAMultiheadAttention_30262339567743
// MI455X (gfx1250) — matrix operand order repaired and checked by reading, not compiled, not run on hardware
//
#include <hip/hip_runtime.h>
#include <hip/hip_bf16.h>
#include <stdint.h>

// ---------------- problem constants ----------------
#define BATCH    4
#define SEQ      2048
#define EMB      512
#define HEADS    8
#define HDIM     64
#define NSPLAT   35
#define NSPLAT_P 64            // padded splat dim for WMMA (35 -> 64, zero pad)
#define BH       (BATCH*HEADS) // 32
#define ROWS     (BATCH*SEQ)   // 8192  (GEMM M)
#define BHS      (BH*SEQ)      // 65536
#define TOPK     64
#define NEG_INF  (-1000000000.0f)

// ---------------- CDNA5 WMMA types ----------------
typedef __attribute__((ext_vector_type(16))) __bf16 v16bf;
typedef __attribute__((ext_vector_type(2)))  __bf16 bf2;
typedef __attribute__((ext_vector_type(8)))  float  v8f;

static __device__ __forceinline__ v8f wmma_bf16(v16bf a, v16bf b, v8f c) {
  // D = A(16x32 bf16) * B(32x16 bf16) + C(16x16 f32)
  return __builtin_amdgcn_wmma_f32_16x16x32_bf16(false, a, false, b, (short)0, c,
                                                 false, false);
}

// A-matrix (16x32, 16-bit) per-lane K offset for VGPR pair g (ISA 7.12.2)
static __device__ __forceinline__ int a_koff(int g, int half) {
  return (g < 4) ? (8 * half + 2 * g) : (16 + 8 * half + 2 * (g - 4));
}

// ---------------- f32 -> bf16 convert ----------------
__global__ void cvt_bf16_kernel(const float* __restrict__ in,
                                __bf16* __restrict__ out, int n) {
  int i = blockIdx.x * blockDim.x + threadIdx.x;
  if (i < n) out[i] = (__bf16)in[i];
}

// ---------------- tiled bf16 GEMM: C[M,N] = X[M,K] @ W[N,K]^T + bias ----------------
// M=8192, N=512, K=512. Block = 256 threads (8 waves), tile 128x64, K-step 32.
// Each wave owns a 2x2 grid of 16x16 output tiles (4 WMMAs per k-step).
// mode 0: Q proj  -> Qf f32 [B,H,S,D], scaled by HDIM^-0.5
// mode 1: V proj  -> Vt bf16 [B,H,D,S]  (transposed so probs@V B-frags load as bf16 pairs)
// mode 2: O proj  -> outF f32 [B,S,E]
__global__ __launch_bounds__(256) void gemm_bf16_kernel(
    const __bf16* __restrict__ X, const __bf16* __restrict__ W,
    const float* __restrict__ bias, float* __restrict__ outF,
    __bf16* __restrict__ outB, int mode) {
  __shared__ __align__(16) __bf16 Xs[128][32];
  __shared__ __align__(16) __bf16 Ws[64][32];

  const int tid  = threadIdx.x;
  const int lane = tid & 31;
  const int wv   = tid >> 5;          // 0..7
  const int half = lane >> 4;
  const int l15  = lane & 15;
  const int m0 = blockIdx.x * 128;
  const int n0 = blockIdx.y * 64;
  const int msubA = (wv & 3) * 2;     // wave owns m-subtiles msubA, msubA+1 (of 8)
  const int npair = wv >> 2;          // wave owns n-subtiles 2*npair, 2*npair+1 (of 4)

  v8f c00 = {}, c01 = {}, c10 = {}, c11 = {};

  const int lr = tid >> 2;            // load row 0..63
  const int lc = (tid & 3) * 8;       // load col group (8 bf16 = 16B)

  for (int k0 = 0; k0 < EMB; k0 += 32) {
    // cooperative tile load
    *(uint4*)&Xs[lr][lc]      = *(const uint4*)&X[(size_t)(m0 + lr) * EMB + k0 + lc];
    *(uint4*)&Xs[lr + 64][lc] = *(const uint4*)&X[(size_t)(m0 + lr + 64) * EMB + k0 + lc];
    *(uint4*)&Ws[lr][lc]      = *(const uint4*)&W[(size_t)(n0 + lr) * EMB + k0 + lc];
    if (k0 + 32 < EMB) {
      __builtin_prefetch(&X[(size_t)(m0 + lr) * EMB + k0 + 32 + lc], 0, 1);
      __builtin_prefetch(&X[(size_t)(m0 + lr + 64) * EMB + k0 + 32 + lc], 0, 1);
      __builtin_prefetch(&W[(size_t)(n0 + lr) * EMB + k0 + 32 + lc], 0, 1);
    }
    __syncthreads();

    v16bf a0, a1;
#pragma unroll
    for (int g = 0; g < 8; ++g) {
      const int kk = a_koff(g, half);
      bf2 p0 = *(const bf2*)&Xs[(msubA + 0) * 16 + l15][kk];
      bf2 p1 = *(const bf2*)&Xs[(msubA + 1) * 16 + l15][kk];
      a0[2 * g] = p0[0]; a0[2 * g + 1] = p0[1];
      a1[2 * g] = p1[0]; a1[2 * g + 1] = p1[1];
    }
    v16bf b0, b1;
#pragma unroll
    for (int v = 0; v < 8; ++v) {
      const int kk = a_koff(v, half);   // B-matrix (32x16) per-lane K offset
      bf2 p0 = *(const bf2*)&Ws[(npair * 2 + 0) * 16 + l15][kk];
      bf2 p1 = *(const bf2*)&Ws[(npair * 2 + 1) * 16 + l15][kk];
      b0[2 * v] = p0[0]; b0[2 * v + 1] = p0[1];
      b1[2 * v] = p1[0]; b1[2 * v + 1] = p1[1];
    }
    c00 = wmma_bf16(a0, b0, c00);
    c01 = wmma_bf16(a0, b1, c01);
    c10 = wmma_bf16(a1, b0, c10);
    c11 = wmma_bf16(a1, b1, c11);
    __syncthreads();
  }

  // epilogue: C layout (16x16 f32): VGPR r, lanes 0-15 -> M=r, lanes 16-31 -> M=8+r; N=lane&15
#pragma unroll
  for (int i = 0; i < 2; ++i) {
#pragma unroll
    for (int r = 0; r < 8; ++r) {
      const int mm = m0 + (msubA + i) * 16 + r + 8 * half;
      const int b  = mm >> 11;          // / SEQ
      const int s  = mm & (SEQ - 1);
#pragma unroll
      for (int t = 0; t < 2; ++t) {
        const v8f& c = (i == 0) ? (t == 0 ? c00 : c01) : (t == 0 ? c10 : c11);
        const int nn  = n0 + (npair * 2 + t) * 16 + l15;
        const float v = c[r] + bias[nn];
        if (mode == 0) {          // Q: scaled, [B,H,S,D] f32
          const int h = nn >> 6, d = nn & 63;
          outF[(size_t)(((b * HEADS + h) * SEQ) + s) * HDIM + d] = v * 0.125f;
        } else if (mode == 1) {   // V: [B,H,D,S] bf16 (transposed)
          const int h = nn >> 6, d = nn & 63;
          outB[(size_t)((b * HEADS + h) * HDIM + d) * SEQ + s] = (__bf16)v;
        } else {                  // O: [B,S,E] f32 (final output)
          outF[(size_t)mm * EMB + nn] = v;
        }
      }
    }
  }
}

// ---------------- splat affinities: affin = exp(-0.5*||q-c||^2) ----------------
__global__ void affin_kernel(const float* __restrict__ Qf,
                             const float* __restrict__ centers,
                             __bf16* __restrict__ Apad,
                             __bf16* __restrict__ Awpad) {
  const int idx = blockIdx.x * blockDim.x + threadIdx.x; // over BHS*64
  if (idx >= BHS * NSPLAT_P) return;
  const int n = idx & 63;
  const int row = idx >> 6;
  float aff = 0.f;
  if (n < NSPLAT) {
    const float* q = Qf + (size_t)row * HDIM;
    const float* c = centers + (size_t)n * HDIM;
    float d2 = 0.f;
#pragma unroll 8
    for (int d = 0; d < HDIM; ++d) { float t = q[d] - c[d]; d2 += t * t; }
    aff = __expf(-0.5f * d2);
  }
  const float w = (n < 20) ? 0.5f : (n < 30) ? 0.3f : (n < NSPLAT) ? 0.2f : 0.f;
  Apad[idx]  = (__bf16)aff;
  Awpad[idx] = (__bf16)(aff * w);
}

// ---------------- fused attention: scores -> exact top-64 -> softmax -> probs@V ----
// 1 workgroup (512 threads, 16 waves) per 16 query rows of one (b,h).
// LDS: Stile[16][2048] f32 (scores; probs rewritten bf16 in place) + Opart[16][256] f32.
__global__ __launch_bounds__(512) void attn_kernel(
    const __bf16* __restrict__ Apad, const __bf16* __restrict__ Awpad,
    const __bf16* __restrict__ Vt, const unsigned char* __restrict__ mask,
    __bf16* __restrict__ Obf) {
  extern __shared__ float smem[];
  float* Stile = smem;                    // [16][SEQ] f32 scores
  float* Opart = smem + 16 * SEQ;         // [16][256]

  const int tid  = threadIdx.x;
  const int lane = tid & 31;
  const int wv   = tid >> 5;              // 0..15
  const int half = lane >> 4;
  const int l15  = lane & 15;

  const int bh = blockIdx.x >> 7;         // 128 row-tiles per (b,h)
  const int i0 = (blockIdx.x & 127) * 16;
  const int b  = bh >> 3;
  const int h  = bh & 7;

  const __bf16* Arow  = Apad  + (size_t)bh * SEQ * NSPLAT_P;
  const __bf16* AWrow = Awpad + (size_t)bh * SEQ * NSPLAT_P;

  // ---- Phase 1: scores[i0..i0+15][0..2047] = (affin*w)[i] . affin[j]  via WMMA ----
  v16bf a0, a1;                           // Aw rows, K=0..31 and K=32..63
#pragma unroll
  for (int g = 0; g < 8; ++g) {
    const int kk = a_koff(g, half);
    bf2 p0 = *(const bf2*)&AWrow[(size_t)(i0 + l15) * NSPLAT_P + kk];
    bf2 p1 = *(const bf2*)&AWrow[(size_t)(i0 + l15) * NSPLAT_P + 32 + kk];
    a0[2 * g] = p0[0]; a0[2 * g + 1] = p0[1];
    a1[2 * g] = p1[0]; a1[2 * g + 1] = p1[1];
  }
  for (int jt = wv; jt < SEQ / 16; jt += 16) {
    const int j0 = jt * 16;
    v16bf b0, b1;                         // B = affin[j]^T : element (k=n, col=j)
#pragma unroll
    for (int v = 0; v < 8; ++v) {
      const int kk = a_koff(v, half);
      bf2 p0 = *(const bf2*)&Arow[(size_t)(j0 + l15) * NSPLAT_P + kk];
      bf2 p1 = *(const bf2*)&Arow[(size_t)(j0 + l15) * NSPLAT_P + 32 + kk];
      b0[2 * v] = p0[0]; b0[2 * v + 1] = p0[1];
      b1[2 * v] = p1[0]; b1[2 * v + 1] = p1[1];
    }
    v8f c = {};
    c = wmma_bf16(a0, b0, c);
    c = wmma_bf16(a1, b1, c);
#pragma unroll
    for (int r = 0; r < 8; ++r)
      Stile[(r + 8 * half) * SEQ + j0 + l15] = c[r];
  }
  __syncthreads();

  // ---- Phase 2: per row (wave wv): exact top-64 threshold, mask, softmax ----
  // Lane owns column pairs j = 64*t + 2*lane + {0,1}. After computing normalized
  // probs, write them back bf16 IN PLACE over the row's own LDS region (safe: all
  // f32 reads complete into registers before any bf16 write; rows are disjoint).
  {
    float* Srow = Stile + (size_t)wv * SEQ;
    float s[64];
#pragma unroll
    for (int t = 0; t < 32; ++t) {
      const float2 ld = *(const float2*)&Srow[64 * t + 2 * lane];
      s[2 * t] = ld.x; s[2 * t + 1] = ld.y;
    }

    // scores are strictly >= 0 -> IEEE bit order == value order: radix select
    unsigned prefix = 0u;
    for (int bit = 30; bit >= 0; --bit) {
      const unsigned cand = prefix | (1u << bit);
      int cnt = 0;
#pragma unroll
      for (int t = 0; t < 64; ++t) cnt += (__float_as_uint(s[t]) >= cand) ? 1 : 0;
#pragma unroll
      for (int off = 16; off > 0; off >>= 1) cnt += __shfl_xor(cnt, off, 32);
      if (cnt >= TOPK) prefix = cand;
    }

    const unsigned char* mrow = mask + (size_t)b * SEQ;
    float mx = NEG_INF;
#pragma unroll
    for (int t = 0; t < 64; ++t) {
      const int j = 64 * (t >> 1) + 2 * lane + (t & 1);
      const bool kept = (__float_as_uint(s[t]) >= prefix);
      const bool msk  = (mrow[j] != 0);
      const float x = (kept && !msk) ? s[t] : NEG_INF;
      s[t] = x;
      mx = fmaxf(mx, x);
    }
#pragma unroll
    for (int off = 16; off > 0; off >>= 1) mx = fmaxf(mx, __shfl_xor(mx, off, 32));

    float sum = 0.f;
#pragma unroll
    for (int t = 0; t < 64; ++t) { float p = __expf(s[t] - mx); s[t] = p; sum += p; }
#pragma unroll
    for (int off = 16; off > 0; off >>= 1) sum += __shfl_xor(sum, off, 32);
    const float inv = 1.f / sum;

    __bf16* Prow = (__bf16*)Srow;         // bf16 probs occupy first half of row region
#pragma unroll
    for (int t = 0; t < 32; ++t) {
      bf2 p;
      p[0] = (__bf16)(s[2 * t] * inv);
      p[1] = (__bf16)(s[2 * t + 1] * inv);
      *(bf2*)&Prow[64 * t + 2 * lane] = p;
    }
  }
  __syncthreads();

  // ---- Phase 3: O[16][64] = P[16][2048] @ V[2048][64] via WMMA ----
  // 16 waves = 4 d-subtiles x 4 K-partitions (512 each). Probs are bf16 in LDS
  // (row r at f32-row base), so A-fragments load as contiguous bf16 pairs.
  {
    const int nt = wv & 3, kp = wv >> 2;
    const int d0 = nt * 16;
    const __bf16* Pbase = (const __bf16*)(Stile + (size_t)l15 * SEQ); // prob row l15
    const __bf16* Vb = Vt + (size_t)bh * HDIM * SEQ;
    v8f c = {};
    for (int ks = 0; ks < 16; ++ks) {
      const int kb = kp * 512 + ks * 32;
      v16bf a, bb;
#pragma unroll
      for (int g = 0; g < 8; ++g) {
        const int kk = kb + a_koff(g, half);
        bf2 p = *(const bf2*)&Pbase[kk];
        a[2 * g] = p[0]; a[2 * g + 1] = p[1];
      }
#pragma unroll
      for (int v = 0; v < 8; ++v) {
        const int kk = kb + a_koff(v, half);   // j index; Vt is [d][j] so pair contiguous
        bf2 p = *(const bf2*)&Vb[(size_t)(d0 + l15) * SEQ + kk];
        bb[2 * v] = p[0]; bb[2 * v + 1] = p[1];
      }
      c = wmma_bf16(a, bb, c);
    }
#pragma unroll
    for (int r = 0; r < 8; ++r)
      Opart[wv * 256 + (r + 8 * half) * 16 + l15] = c[r];
  }
  __syncthreads();

  // reduce K-partitions, write O (bf16, [B,S,E] with e = h*64+d) for the final GEMM
  for (int o = tid; o < 16 * HDIM; o += 512) {
    const int m = o >> 6, d = o & 63;
    const int nt2 = d >> 4, nc = d & 15;
    float acc = 0.f;
#pragma unroll
    for (int kp2 = 0; kp2 < 4; ++kp2)
      acc += Opart[(kp2 * 4 + nt2) * 256 + m * 16 + nc];
    Obf[(size_t)(b * SEQ + i0 + m) * EMB + h * HDIM + d] = (__bf16)acc;
  }
}

// ---------------- host-side launch ----------------
extern "C" void kernel_launch(void* const* d_in, const int* in_sizes, int n_in,
                              void* d_out, int out_size, void* d_ws, size_t ws_size,
                              hipStream_t stream) {
  const float* query   = (const float*)d_in[0];
  // d_in[1] = key (projection result is discarded by the reference -> skipped)
  const float* value   = (const float*)d_in[2];
  const unsigned char* mask = (const unsigned char*)d_in[3];
  const float* Wq = (const float*)d_in[4];
  const float* bq = (const float*)d_in[5];
  const float* Wv = (const float*)d_in[8];
  const float* bv = (const float*)d_in[9];
  const float* Wo = (const float*)d_in[10];
  const float* bo = (const float*)d_in[11];
  const float* centers = (const float*)d_in[12];

  char* ws = (char*)d_ws;
  size_t off = 0;
  auto alloc = [&](size_t bytes) {
    char* p = ws + off;
    off += (bytes + 255) & ~(size_t)255;
    return p;
  };
  __bf16* Xq  = (__bf16*)alloc((size_t)ROWS * EMB * 2);   // query bf16
  __bf16* Xv  = (__bf16*)alloc((size_t)ROWS * EMB * 2);   // value bf16
  __bf16* Wqb = (__bf16*)alloc((size_t)EMB * EMB * 2);
  __bf16* Wvb = (__bf16*)alloc((size_t)EMB * EMB * 2);
  __bf16* Wob = (__bf16*)alloc((size_t)EMB * EMB * 2);
  float*  Qf  = (float*) alloc((size_t)BHS * HDIM * 4);   // scaled q, [B,H,S,D]
  __bf16* Vt  = (__bf16*)alloc((size_t)BHS * HDIM * 2);   // [B,H,D,S]
  __bf16* Apad  = (__bf16*)alloc((size_t)BHS * NSPLAT_P * 2);
  __bf16* Awpad = (__bf16*)alloc((size_t)BHS * NSPLAT_P * 2);
  __bf16* Obf = (__bf16*)alloc((size_t)ROWS * EMB * 2);   // attn out, [B,S,E]

  const int nx = ROWS * EMB, nw = EMB * EMB;
  cvt_bf16_kernel<<<(nx + 255) / 256, 256, 0, stream>>>(query, Xq, nx);
  cvt_bf16_kernel<<<(nx + 255) / 256, 256, 0, stream>>>(value, Xv, nx);
  cvt_bf16_kernel<<<(nw + 255) / 256, 256, 0, stream>>>(Wq, Wqb, nw);
  cvt_bf16_kernel<<<(nw + 255) / 256, 256, 0, stream>>>(Wv, Wvb, nw);
  cvt_bf16_kernel<<<(nw + 255) / 256, 256, 0, stream>>>(Wo, Wob, nw);

  dim3 ggrid(ROWS / 128, EMB / 64);
  gemm_bf16_kernel<<<ggrid, 256, 0, stream>>>(Xq, Wqb, bq, Qf, nullptr, 0);
  gemm_bf16_kernel<<<ggrid, 256, 0, stream>>>(Xv, Wvb, bv, nullptr, Vt, 1);

  const int na = BHS * NSPLAT_P;
  affin_kernel<<<(na + 255) / 256, 256, 0, stream>>>(Qf, centers, Apad, Awpad);

  const size_t attn_lds = (size_t)(16 * SEQ + 16 * 256) * sizeof(float); // 144 KB
  attn_kernel<<<BH * (SEQ / 16), 512, attn_lds, stream>>>(Apad, Awpad, Vt, mask, Obf);

  gemm_bf16_kernel<<<ggrid, 256, 0, stream>>>(Obf, Wob, bo, (float*)d_out, nullptr, 2);
}
